// ExponentialStateMixer_40922448396563
// MI455X (gfx1250) — matrix operand order repaired and checked by reading, not compiled, not run on hardware
//
#include <hip/hip_runtime.h>
#include <hip/hip_bf16.h>

// ---------------------------------------------------------------------------
// ExponentialStateMixer on MI455X (gfx1250, wave32, WMMA)
//   h = x @ W_in + b_in ; alpha = sigmoid(gate); c = tanh(cand)
//   y_t = (1-alpha_t) y_{t-1} + alpha_t c_t  (scan over T)
//   out = y @ W_out + b_out
// GEMMs: v_wmma_f32_16x16x32_f16, f32 accumulate.
// Scan: 3-phase chunked parallel scan (latency-hiding, bandwidth-bound).
// Intermediates kept f16 in workspace to halve stream traffic.
// ---------------------------------------------------------------------------

typedef __attribute__((ext_vector_type(16))) _Float16 v16h;
typedef __attribute__((ext_vector_type(8)))  _Float16 v8h;
typedef __attribute__((ext_vector_type(4)))  _Float16 v4h;
typedef __attribute__((ext_vector_type(2)))  _Float16 v2h;
typedef __attribute__((ext_vector_type(8)))  float    v8f;

namespace esm {
constexpr int Bn = 8;
constexpr int Tn = 2048;
constexpr int Sn = 8;
constexpr int Dn = 256;
constexpr int Rrows  = Bn * Tn * Sn;       // 131072 flattened tokens
constexpr int NCH    = Bn * Sn * Dn;       // 16384 scan channels
constexpr int CHUNKS = 32;                 // T chunks for parallel scan
constexpr int CLEN   = Tn / CHUNKS;        // 64 steps per chunk
constexpr int LDS_STR = Dn + 8;            // padded f16 row stride (bank-conflict-free)
}

// ---------------------------------------------------------------------------
// Kernel 0: convert + transpose weights to f16 column-major.
//   Wt_in [e][k] = W_in[k][e]   (e<512, k<256)
//   Wt_out[e][k] = W_out[k][e]  (e<256, k<256)
// ---------------------------------------------------------------------------
__global__ void esm_cvt_weights(const float* __restrict__ W_in,
                                const float* __restrict__ W_out,
                                _Float16* __restrict__ Wt_in,
                                _Float16* __restrict__ Wt_out) {
  int tid = blockIdx.x * blockDim.x + threadIdx.x;
  if (tid < 512 * 256) {
    int e = tid >> 8, k = tid & 255;
    Wt_in[tid] = (_Float16)W_in[k * 512 + e];
  } else {
    int o = tid - 512 * 256;
    if (o < 256 * 256) {
      int e = o >> 8, k = o & 255;
      Wt_out[o] = (_Float16)W_out[k * 256 + e];
    }
  }
}

// ---------------------------------------------------------------------------
// WMMA fragment helpers (layouts per CDNA5 ISA 7.12.2, wave32)
//   A 16x32 f16 : lane(l<16) row m=l, K runs {k0..k0+7} and {k0+16..k0+23};
//                 lanes 16-31 same rows, runs shifted by +8.
//   B 32x16 f16 : lane col n=l&15, same K runs as A (k0+half*8.., +16).
//   C/D 16x16 f32: lane col n=l&15, vgpr r -> row m = r + 8*(l>>4).
// ---------------------------------------------------------------------------
union FragH { v16h v; v8h h[2]; };

__device__ inline v16h load_fragA_lds(const _Float16* rowPtr, int kc, int laneHalf) {
  FragH f;
  int base = kc * 32 + laneHalf * 8;
  f.h[0] = *(const v8h*)(rowPtr + base);
  f.h[1] = *(const v8h*)(rowPtr + base + 16);
  return f.v;
}

__device__ inline v16h load_fragB_glb(const _Float16* colPtr, int kc, int laneHalf) {
  FragH f;
  int base = kc * 32 + laneHalf * 8;
  f.h[0] = *(const v8h*)(colPtr + base);
  f.h[1] = *(const v8h*)(colPtr + base + 16);
  return f.v;
}

// ---------------------------------------------------------------------------
// Kernel 1: in_proj + gating.  grid = (R/16, 2), block = 256 (8 waves).
// Block: 16 token rows; wave w owns the gate tile (col n) and cand tile
// (col n+256) so the sigmoid/tanh epilogue fuses into a/b emission.
// ---------------------------------------------------------------------------
__global__ __launch_bounds__(256)
void esm_inproj(const float* __restrict__ x,
                const _Float16* __restrict__ Wt_in,   // [512][256] col-major f16
                const float* __restrict__ b_in,
                _Float16* __restrict__ a_out,         // [R][256] f16
                _Float16* __restrict__ bv_out) {      // [R][256] f16
  using namespace esm;
  __shared__ _Float16 ldsA[16 * LDS_STR];
  const int row0 = blockIdx.x * 16;

  // Stage 16x256 f32 x-tile -> f16 LDS (coalesced float4 loads).
#pragma unroll
  for (int it = 0; it < 4; ++it) {
    int g  = threadIdx.x + it * 256;       // 0..1023 float4 groups
    int r  = g >> 6;                       // row 0..15
    int c4 = (g & 63) << 2;                // col 0..252 step 4
    float4 v = *(const float4*)(x + (size_t)(row0 + r) * Dn + c4);
    v4h hv = { (_Float16)v.x, (_Float16)v.y, (_Float16)v.z, (_Float16)v.w };
    *(v4h*)&ldsA[r * LDS_STR + c4] = hv;
  }
  __syncthreads();

  const int lane     = threadIdx.x & 31;
  const int wave     = threadIdx.x >> 5;
  const int laneHalf = lane >> 4;
  const int lo       = lane & 15;
  const int n        = blockIdx.y * 128 + wave * 16 + lo;  // gate column

  v8f accG, accC;
  const float bg = b_in[n];
  const float bc = b_in[n + 256];
#pragma unroll
  for (int i = 0; i < 8; ++i) { accG[i] = bg; accC[i] = bc; }

  const _Float16* Arow = &ldsA[lo * LDS_STR];
  const _Float16* Bg   = Wt_in + (size_t)n * Dn;
  const _Float16* Bc   = Wt_in + (size_t)(n + 256) * Dn;

#pragma unroll
  for (int kc = 0; kc < 8; ++kc) {
    v16h fa = load_fragA_lds(Arow, kc, laneHalf);
    v16h fg = load_fragB_glb(Bg,  kc, laneHalf);
    v16h fc = load_fragB_glb(Bc,  kc, laneHalf);
    accG = __builtin_amdgcn_wmma_f32_16x16x32_f16(false, fa, false, fg,
                                                  (short)0, accG, false, false);
    accC = __builtin_amdgcn_wmma_f32_16x16x32_f16(false, fa, false, fc,
                                                  (short)0, accC, false, false);
  }

  // Fused activation epilogue: a = 1-sigmoid(g), b = sigmoid(g)*tanh(c)
#pragma unroll
  for (int r = 0; r < 8; ++r) {
    int m = r + laneHalf * 8;
    size_t off = (size_t)(row0 + m) * Dn + n;
    float alpha = 1.0f / (1.0f + __expf(-accG[r]));
    float cnd   = tanhf(accC[r]);
    a_out[off]  = (_Float16)(1.0f - alpha);
    bv_out[off] = (_Float16)(alpha * cnd);
  }
}

// ---------------------------------------------------------------------------
// Scan phase 1: per-(channel-pair, chunk) affine aggregate.
// Each thread owns two adjacent d-channels (paired v2h loads -> each wave
// reads one full contiguous 128B line per step). 262144 threads.
//   state_out = A * state_in + B, with A = prod(a_t), B = scan from 0.
// ---------------------------------------------------------------------------
__global__ __launch_bounds__(256)
void esm_scan_agg(const _Float16* __restrict__ a_h,
                  const _Float16* __restrict__ b_h,
                  float* __restrict__ aggA,      // [CHUNKS][NCH]
                  float* __restrict__ aggB) {    // [CHUNKS][NCH]
  using namespace esm;
  const int tid = blockIdx.x * blockDim.x + threadIdx.x;  // 0 .. 32*8192-1
  const int p  = tid & 8191;          // channel-pair index
  const int c  = tid >> 13;           // chunk
  const int d  = (p & 127) << 1;      // paired d
  const int s  = (p >> 7) & (Sn - 1);
  const int b  = p >> 10;
  size_t idx = (((size_t)b * Tn + c * CLEN) * Sn + s) * Dn + d;
  const size_t stride = (size_t)Sn * Dn;                  // 2048 elems / step

  float A0 = 1.0f, B0 = 0.0f, A1 = 1.0f, B1 = 0.0f;
#pragma unroll 4
  for (int t = 0; t < CLEN; ++t) {
    __builtin_prefetch(a_h + idx + 8 * stride, 0, 1);     // global_prefetch_b8
    __builtin_prefetch(b_h + idx + 8 * stride, 0, 1);
    v2h av = *(const v2h*)(a_h + idx);
    v2h bv = *(const v2h*)(b_h + idx);
    float a0 = (float)av[0], a1 = (float)av[1];
    A0 *= a0;  B0 = a0 * B0 + (float)bv[0];
    A1 *= a1;  B1 = a1 * B1 + (float)bv[1];
    idx += stride;
  }
  const int ch = ((b * Sn + s) << 8) + d;                 // flat channel
  *(float2*)(aggA + (size_t)c * NCH + ch) = make_float2(A0, A1);
  *(float2*)(aggB + (size_t)c * NCH + ch) = make_float2(B0, B1);
}

// ---------------------------------------------------------------------------
// Scan phase 2: per-channel exclusive scan over the 32 chunk aggregates.
// 16384 threads, 32 tiny steps, fully coalesced ([c][ch] layout).
// ---------------------------------------------------------------------------
__global__ __launch_bounds__(256)
void esm_scan_carry(const float* __restrict__ aggA,
                    const float* __restrict__ aggB,
                    float* __restrict__ pref) {           // [CHUNKS][NCH]
  using namespace esm;
  const int ch = blockIdx.x * blockDim.x + threadIdx.x;   // 0..16383
  float carry = 0.0f;
#pragma unroll
  for (int c = 0; c < CHUNKS; ++c) {
    size_t o = (size_t)c * NCH + ch;
    pref[o] = carry;
    carry = aggA[o] * carry + aggB[o];
  }
}

// ---------------------------------------------------------------------------
// Scan phase 3: re-run recurrence per chunk seeded with its carry-in,
// write y (f16) exactly once. Same mapping/coalescing as phase 1.
// ---------------------------------------------------------------------------
__global__ __launch_bounds__(256)
void esm_scan_apply(const _Float16* __restrict__ a_h,
                    const _Float16* __restrict__ b_h,
                    const float* __restrict__ pref,
                    _Float16* __restrict__ y_h) {
  using namespace esm;
  const int tid = blockIdx.x * blockDim.x + threadIdx.x;
  const int p  = tid & 8191;
  const int c  = tid >> 13;
  const int d  = (p & 127) << 1;
  const int s  = (p >> 7) & (Sn - 1);
  const int b  = p >> 10;
  size_t idx = (((size_t)b * Tn + c * CLEN) * Sn + s) * Dn + d;
  const size_t stride = (size_t)Sn * Dn;

  const int ch = ((b * Sn + s) << 8) + d;
  float2 seed = *(const float2*)(pref + (size_t)c * NCH + ch);
  float y0 = seed.x, y1 = seed.y;
#pragma unroll 4
  for (int t = 0; t < CLEN; ++t) {
    __builtin_prefetch(a_h + idx + 8 * stride, 0, 1);
    __builtin_prefetch(b_h + idx + 8 * stride, 0, 1);
    v2h av = *(const v2h*)(a_h + idx);
    v2h bv = *(const v2h*)(b_h + idx);
    y0 = (float)av[0] * y0 + (float)bv[0];
    y1 = (float)av[1] * y1 + (float)bv[1];
    v2h yv = { (_Float16)y0, (_Float16)y1 };
    *(v2h*)(y_h + idx) = yv;
    idx += stride;
  }
}

// ---------------------------------------------------------------------------
// Kernel 3: out_proj. grid = (R/16, 2), block = 256 (8 waves).
// ---------------------------------------------------------------------------
__global__ __launch_bounds__(256)
void esm_outproj(const _Float16* __restrict__ y_h,      // [R][256] f16
                 const _Float16* __restrict__ Wt_out,   // [256][256] col-major f16
                 const float* __restrict__ b_out,
                 float* __restrict__ out) {             // [R][256] f32
  using namespace esm;
  __shared__ _Float16 ldsA[16 * LDS_STR];
  const int row0 = blockIdx.x * 16;

  // Stage 16x256 f16 y-tile into padded LDS (128-bit loads/stores).
#pragma unroll
  for (int it = 0; it < 2; ++it) {
    int g  = threadIdx.x + it * 256;       // 0..511 v8h groups
    int r  = g >> 5;                       // row 0..15
    int c8 = (g & 31) << 3;                // col 0..248 step 8
    v8h v = *(const v8h*)(y_h + (size_t)(row0 + r) * Dn + c8);
    *(v8h*)&ldsA[r * LDS_STR + c8] = v;
  }
  __syncthreads();

  const int lane     = threadIdx.x & 31;
  const int wave     = threadIdx.x >> 5;
  const int laneHalf = lane >> 4;
  const int lo       = lane & 15;
  const int n        = blockIdx.y * 128 + wave * 16 + lo;

  v8f acc;
  const float bo = b_out[n];
#pragma unroll
  for (int i = 0; i < 8; ++i) acc[i] = bo;

  const _Float16* Arow = &ldsA[lo * LDS_STR];
  const _Float16* Bcol = Wt_out + (size_t)n * Dn;

#pragma unroll
  for (int kc = 0; kc < 8; ++kc) {
    v16h fa = load_fragA_lds(Arow, kc, laneHalf);
    v16h fb = load_fragB_glb(Bcol, kc, laneHalf);
    acc = __builtin_amdgcn_wmma_f32_16x16x32_f16(false, fa, false, fb,
                                                 (short)0, acc, false, false);
  }

#pragma unroll
  for (int r = 0; r < 8; ++r) {
    int m = r + laneHalf * 8;
    out[(size_t)(row0 + m) * Dn + n] = acc[r];
  }
}

// ---------------------------------------------------------------------------
// Host-side launch. Inputs: x, W_in, b_in, W_out, b_out (all f32).
// WS: Wt_in 256KB | Wt_out 128KB | a 64MB | b 64MB | y 64MB | aggA/aggB/pref 2MB ea
// ---------------------------------------------------------------------------
extern "C" void kernel_launch(void* const* d_in, const int* in_sizes, int n_in,
                              void* d_out, int out_size, void* d_ws, size_t ws_size,
                              hipStream_t stream) {
  using namespace esm;
  const float* x     = (const float*)d_in[0];
  const float* W_in  = (const float*)d_in[1];
  const float* b_in  = (const float*)d_in[2];
  const float* W_out = (const float*)d_in[3];
  const float* b_out = (const float*)d_in[4];
  float* out = (float*)d_out;

  char* ws = (char*)d_ws;
  _Float16* Wt_in  = (_Float16*)ws;                                   // 512*256
  _Float16* Wt_out = (_Float16*)(ws + (size_t)512 * 256 * 2);         // 256*256
  _Float16* a_h    = (_Float16*)(ws + (size_t)512 * 256 * 2
                                    + (size_t)256 * 256 * 2);
  _Float16* bv_h   = a_h + (size_t)Rrows * Dn;
  _Float16* y_h    = bv_h + (size_t)Rrows * Dn;
  float*    aggA   = (float*)(y_h + (size_t)Rrows * Dn);
  float*    aggB   = aggA + (size_t)CHUNKS * NCH;
  float*    pref   = aggB + (size_t)CHUNKS * NCH;

  // 0) weight convert/transpose: 512*256 + 256*256 = 196608 elements
  esm_cvt_weights<<<768, 256, 0, stream>>>(W_in, W_out, Wt_in, Wt_out);

  // 1) in_proj + gating
  dim3 g1(Rrows / 16, 2);
  esm_inproj<<<g1, 256, 0, stream>>>(x, Wt_in, b_in, a_h, bv_h);

  // 2) chunked parallel scan over T
  const int scanThreads = (NCH / 2) * CHUNKS;             // 262144
  esm_scan_agg  <<<scanThreads / 256, 256, 0, stream>>>(a_h, bv_h, aggA, aggB);
  esm_scan_carry<<<NCH / 256,        256, 0, stream>>>(aggA, aggB, pref);
  esm_scan_apply<<<scanThreads / 256, 256, 0, stream>>>(a_h, bv_h, pref, y_h);

  // 3) out_proj
  dim3 g3(Rrows / 16, 2);
  esm_outproj<<<g3, 256, 0, stream>>>(y_h, Wt_out, b_out, out);
}
